// AdvancedGNN_61168924230300
// MI455X (gfx1250) — matrix operand order repaired and checked by reading, not compiled, not run on hardware
//
#include <hip/hip_runtime.h>
#include <hip/hip_bf16.h>

typedef __attribute__((ext_vector_type(16))) _Float16 v16h;
typedef __attribute__((ext_vector_type(8)))  _Float16 v8h;
typedef __attribute__((ext_vector_type(8)))  float    v8f;

static constexpr int NND = 20000;   // nodes
static constexpr int NED = 320000;  // edges
static constexpr int FIN = 256;     // input feature dim
static constexpr int D1  = 512;     // hid*heads layer1
static constexpr int H1N = 4;       // heads layer1
static constexpr int C1  = 128;     // per-head channels
static constexpr int OC  = 128;     // out channels (layer2, H=1)
static constexpr int EDM = 26;      // edge feature dim

// ---- float <-> orderable uint (for atomic float max via atomicMax(u32)) ----
static __device__ __forceinline__ unsigned f2o(float f) {
  unsigned u = __float_as_uint(f);
  return (u & 0x80000000u) ? ~u : (u | 0x80000000u);
}
static __device__ __forceinline__ float o2f(unsigned u) {
  return __uint_as_float((u & 0x80000000u) ? (u & 0x7FFFFFFFu) : ~u);
}

// ---------------- small utility kernels ----------------
__global__ void k_cvt16(const float* __restrict__ in, _Float16* __restrict__ out, int n) {
  int i = blockIdx.x * blockDim.x + threadIdx.x;
  if (i < n) out[i] = (_Float16)in[i];
}

// W [K x N] row-major f32  ->  Wt [N x K] row-major f16
__global__ void k_wT16(const float* __restrict__ W, _Float16* __restrict__ Wt, int K, int N) {
  int i = blockIdx.x * blockDim.x + threadIdx.x;
  if (i >= N * K) return;
  int n = i / K, k = i - n * K;
  Wt[i] = (_Float16)W[(size_t)k * N + n];
}

__global__ void k_vadd(const float* __restrict__ a, const float* __restrict__ b,
                       float* __restrict__ c, int n) {
  int i = blockIdx.x * blockDim.x + threadIdx.x;
  if (i < n) c[i] = a[i] + b[i];
}

__global__ void k_fill_u32(unsigned* __restrict__ p, unsigned v, int n) {
  int i = blockIdx.x * blockDim.x + threadIdx.x;
  if (i < n) p[i] = v;
}
__global__ void k_fill_f32(float* __restrict__ p, float v, int n) {
  int i = blockIdx.x * blockDim.x + threadIdx.x;
  if (i < n) p[i] = v;
}

// ------------- WMMA GEMM, 2x2 register-blocked: C[MxN] = A16[MxK] * Bt16[NxK]^T + bias
// One wave computes a 32x32 macro-tile (4 x 16x16 WMMA accumulators).
// 8 b128 fragment loads feed 4 WMMAs per K-step (2x better reuse than 1-tile/wave).
// K multiple of 32; M,N multiples of 32.
__global__ void k_wmma_gemm(const _Float16* __restrict__ A, const _Float16* __restrict__ Bt,
                            const float* __restrict__ bias, float* __restrict__ C,
                            int M, int N, int K, int nTiles) {
  int wid = blockIdx.x * (blockDim.x >> 5) + (threadIdx.x >> 5);
  if (wid >= nTiles) return;
  int lane = threadIdx.x & 31;
  int tilesN = N >> 5;                       // macro tiles of 32 columns
  int tm = wid / tilesN, tn = wid - tm * tilesN;
  int half = lane >> 4, l15 = lane & 15;

  const _Float16* a0p = A + (size_t)(tm * 32 + l15) * K;       // A rows m .. (frag 0)
  const _Float16* a1p = a0p + (size_t)16 * K;                  // A rows m+16 (frag 1)
  const _Float16* b0p = Bt + (size_t)(tn * 32 + l15) * K;      // B cols n .. (frag 0)
  const _Float16* b1p = b0p + (size_t)16 * K;                  // B cols n+16 (frag 1)

  v8f acc00 = {}, acc01 = {}, acc10 = {}, acc11 = {};
  for (int k0 = 0; k0 < K; k0 += 32) {
    __builtin_prefetch((const void*)(a0p + k0 + 128), 0, 1);
    __builtin_prefetch((const void*)(b0p + k0 + 128), 0, 1);
    // A 16x32 f16 ISA layout: v16h[i] holds K = k0 + i + (i>=8 ? 8 : 0) + half*8
    v8h aa0 = *(const v8h*)(a0p + k0 + half * 8);
    v8h aa1 = *(const v8h*)(a0p + k0 + 16 + half * 8);
    v8h ab0 = *(const v8h*)(a1p + k0 + half * 8);
    v8h ab1 = *(const v8h*)(a1p + k0 + 16 + half * 8);
    // B 32x16 f16 ISA layout: v16h[i] holds K = k0 + i + (i>=8 ? 8 : 0) + half*8
    v8h ba0 = *(const v8h*)(b0p + k0 + half * 8);
    v8h ba1 = *(const v8h*)(b0p + k0 + 16 + half * 8);
    v8h bb0 = *(const v8h*)(b1p + k0 + half * 8);
    v8h bb1 = *(const v8h*)(b1p + k0 + 16 + half * 8);
    v16h af0, af1, bf0, bf1;
#pragma unroll
    for (int i = 0; i < 8; ++i) {
      af0[i] = aa0[i]; af0[i + 8] = aa1[i];
      af1[i] = ab0[i]; af1[i + 8] = ab1[i];
      bf0[i] = ba0[i]; bf0[i + 8] = ba1[i];
      bf1[i] = bb0[i]; bf1[i + 8] = bb1[i];
    }
    acc00 = __builtin_amdgcn_wmma_f32_16x16x32_f16(false, af0, false, bf0, (short)0, acc00,
                                                   false, false);
    acc01 = __builtin_amdgcn_wmma_f32_16x16x32_f16(false, af0, false, bf1, (short)0, acc01,
                                                   false, false);
    acc10 = __builtin_amdgcn_wmma_f32_16x16x32_f16(false, af1, false, bf0, (short)0, acc10,
                                                   false, false);
    acc11 = __builtin_amdgcn_wmma_f32_16x16x32_f16(false, af1, false, bf1, (short)0, acc11,
                                                   false, false);
  }
  int n0 = tn * 32 + l15, n1 = n0 + 16;
  float bv0 = bias ? bias[n0] : 0.0f;
  float bv1 = bias ? bias[n1] : 0.0f;
#pragma unroll
  for (int v = 0; v < 8; ++v) {
    int m0 = tm * 32 + v + half * 8;          // C/D layout: VGPR v -> M = v + half*8
    int m1 = m0 + 16;
    C[(size_t)m0 * N + n0] = acc00[v] + bv0;
    C[(size_t)m0 * N + n1] = acc01[v] + bv1;
    C[(size_t)m1 * N + n0] = acc10[v] + bv0;
    C[(size_t)m1 * N + n1] = acc11[v] + bv1;
  }
}

// ---------------- edge pass 1: attention logits + segment max ----------------
// wave per edge; computes edge_attr@We on the fly (26-term FMA per element)
__global__ void k_edge_alpha(const float* __restrict__ xl, const float* __restrict__ xr,
                             const float* __restrict__ ea, const float* __restrict__ We,
                             const float* __restrict__ att,
                             const int* __restrict__ src, const int* __restrict__ dst,
                             float* __restrict__ alpha, unsigned* __restrict__ amax,
                             int E, int H, int C) {
  int e = blockIdx.x * (blockDim.x >> 5) + (threadIdx.x >> 5);
  if (e >= E) return;
  int lane = threadIdx.x & 31;
  int D = H * C;
  int s = src[e], d = dst[e];
  const float* aev = ea + (size_t)e * EDM;
  float eav[EDM];
#pragma unroll
  for (int k = 0; k < EDM; ++k) eav[k] = aev[k];

  for (int h = 0; h < H; ++h) {
    float acc = 0.0f;
    for (int t = h * C + lane; t < (h + 1) * C; t += 32) {
      float ecol = 0.0f;
#pragma unroll
      for (int k = 0; k < EDM; ++k) ecol = fmaf(eav[k], We[(size_t)k * D + t], ecol);
      float m = xl[(size_t)s * D + t] + xr[(size_t)d * D + t] + ecol;
      m = m > 0.0f ? m : 0.2f * m;  // leaky_relu(0.2)
      acc = fmaf(m, att[t], acc);
    }
#pragma unroll
    for (int o = 16; o > 0; o >>= 1) acc += __shfl_xor(acc, o, 32);
    if (lane == 0) {
      alpha[(size_t)e * H + h] = acc;
      atomicMax(&amax[(size_t)d * H + h], f2o(acc));
    }
  }
}

// ---------------- edge pass 2: exp + segment sum ----------------
__global__ void k_edge_exp(const float* __restrict__ alpha, const unsigned* __restrict__ amax,
                           const int* __restrict__ dst, float* __restrict__ ex,
                           float* __restrict__ den, int E, int H) {
  int i = blockIdx.x * blockDim.x + threadIdx.x;
  if (i >= E * H) return;
  int e = i / H, h = i - e * H;
  int d = dst[e];
  float v = __expf(alpha[i] - o2f(amax[(size_t)d * H + h]));
  ex[i] = v;
  atomicAdd(&den[(size_t)d * H + h], v);
}

// ---------------- edge pass 3: weighted scatter-add ----------------
__global__ void k_edge_scatter(const float* __restrict__ xl, const float* __restrict__ ex,
                               const float* __restrict__ den,
                               const int* __restrict__ src, const int* __restrict__ dst,
                               float* __restrict__ out, int E, int H, int C) {
  int e = blockIdx.x * (blockDim.x >> 5) + (threadIdx.x >> 5);
  if (e >= E) return;
  int lane = threadIdx.x & 31;
  int D = H * C;
  int s = src[e], d = dst[e];
  for (int h = 0; h < H; ++h) {
    float a = ex[(size_t)e * H + h] / (den[(size_t)d * H + h] + 1e-16f);
    for (int t = h * C + lane; t < (h + 1) * C; t += 32)
      atomicAdd(&out[(size_t)d * D + t], xl[(size_t)s * D + t] * a);
  }
}

// ---------------- LayerNorm(+ELU) row kernels, wave per row ----------------
__global__ void k_ln_elu_h16(const float* __restrict__ in, const float* __restrict__ g,
                             const float* __restrict__ b, _Float16* __restrict__ h16,
                             int rows, int D) {
  int row = blockIdx.x * (blockDim.x >> 5) + (threadIdx.x >> 5);
  if (row >= rows) return;
  int lane = threadIdx.x & 31;
  const float* x = in + (size_t)row * D;
  float s = 0.0f, s2 = 0.0f;
  for (int t = lane; t < D; t += 32) { float v = x[t]; s += v; s2 += v * v; }
#pragma unroll
  for (int o = 16; o > 0; o >>= 1) { s += __shfl_xor(s, o, 32); s2 += __shfl_xor(s2, o, 32); }
  float mu = s / (float)D;
  float inv = rsqrtf(s2 / (float)D - mu * mu + 1e-5f);
  for (int t = lane; t < D; t += 32) {
    float v = (x[t] - mu) * inv * g[t] + b[t];
    v = v > 0.0f ? v : (__expf(v) - 1.0f);  // ELU
    h16[(size_t)row * D + t] = (_Float16)v;
  }
}

__global__ void k_ln_out(const float* __restrict__ in, const float* __restrict__ g,
                         const float* __restrict__ b, float* __restrict__ out,
                         int rows, int D) {
  int row = blockIdx.x * (blockDim.x >> 5) + (threadIdx.x >> 5);
  if (row >= rows) return;
  int lane = threadIdx.x & 31;
  const float* x = in + (size_t)row * D;
  float s = 0.0f, s2 = 0.0f;
  for (int t = lane; t < D; t += 32) { float v = x[t]; s += v; s2 += v * v; }
#pragma unroll
  for (int o = 16; o > 0; o >>= 1) { s += __shfl_xor(s, o, 32); s2 += __shfl_xor(s2, o, 32); }
  float mu = s / (float)D;
  float inv = rsqrtf(s2 / (float)D - mu * mu + 1e-5f);
  for (int t = lane; t < D; t += 32)
    out[(size_t)row * D + t] = (x[t] - mu) * inv * g[t] + b[t];
}

// ---------------- host launcher ----------------
extern "C" void kernel_launch(void* const* d_in, const int* in_sizes, int n_in,
                              void* d_out, int out_size, void* d_ws, size_t ws_size,
                              hipStream_t stream) {
  (void)in_sizes; (void)n_in; (void)out_size; (void)ws_size;

  const float* x     = (const float*)d_in[0];
  const int*   ei    = (const int*)d_in[1];
  const float* ea    = (const float*)d_in[2];
  const float* W1l   = (const float*)d_in[3];
  const float* b1l   = (const float*)d_in[4];
  const float* W1r   = (const float*)d_in[5];
  const float* b1r   = (const float*)d_in[6];
  const float* We1   = (const float*)d_in[7];
  const float* att1  = (const float*)d_in[8];
  const float* bias1 = (const float*)d_in[9];
  const float* Ws1   = (const float*)d_in[10];
  const float* bs1   = (const float*)d_in[11];
  const float* g1    = (const float*)d_in[12];
  const float* be1   = (const float*)d_in[13];
  const float* W2l   = (const float*)d_in[14];
  const float* b2l   = (const float*)d_in[15];
  const float* W2r   = (const float*)d_in[16];
  const float* b2r   = (const float*)d_in[17];
  const float* We2   = (const float*)d_in[18];
  const float* att2  = (const float*)d_in[19];
  const float* bias2 = (const float*)d_in[20];
  const float* Ws2   = (const float*)d_in[21];
  const float* bs2   = (const float*)d_in[22];
  const float* g2    = (const float*)d_in[23];
  const float* be2   = (const float*)d_in[24];

  const int* src = ei;
  const int* dst = ei + NED;

  // ---- workspace carve-out (deterministic) ----
  char* w = (char*)d_ws;
  size_t off = 0;
  auto take = [&](size_t bytes) -> char* {
    char* p = w + off;
    off = (off + bytes + 255) & ~(size_t)255;
    return p;
  };
  _Float16* x16  = (_Float16*)take((size_t)NND * FIN * 2);
  _Float16* h16  = (_Float16*)take((size_t)NND * D1 * 2);
  _Float16* W1lT = (_Float16*)take((size_t)FIN * D1 * 2);
  _Float16* W1rT = (_Float16*)take((size_t)FIN * D1 * 2);
  _Float16* Ws1T = (_Float16*)take((size_t)FIN * D1 * 2);
  _Float16* W2lT = (_Float16*)take((size_t)D1 * OC * 2);
  _Float16* W2rT = (_Float16*)take((size_t)D1 * OC * 2);
  _Float16* Ws2T = (_Float16*)take((size_t)D1 * OC * 2);
  float*    cb1  = (float*)take((size_t)D1 * 4);
  float*    cb2  = (float*)take((size_t)OC * 4);
  float*    xl1  = (float*)take((size_t)NND * D1 * 4);
  float*    xr1  = (float*)take((size_t)NND * D1 * 4);
  float*    gat1 = (float*)take((size_t)NND * D1 * 4);
  float*    alph = (float*)take((size_t)NED * H1N * 4);
  float*    ex   = (float*)take((size_t)NED * H1N * 4);
  unsigned* amax = (unsigned*)take((size_t)NND * H1N * 4);
  float*    den  = (float*)take((size_t)NND * H1N * 4);
  // layer-2 buffers alias layer-1 (stream-ordered, layer-1 consumers all finished)
  float* xl2 = xl1;
  float* xr2 = xr1;
  float* gat2 = gat1;

  auto cdiv = [](long a, long b) { return (int)((a + b - 1) / b); };
  const int T = 256;
  const unsigned NEG_INF_ORD = 0x007FFFFFu;  // f2o(-inf)

  // conversions / weight transposes
  k_cvt16<<<cdiv((long)NND * FIN, T), T, 0, stream>>>(x, x16, NND * FIN);
  k_wT16<<<cdiv((long)FIN * D1, T), T, 0, stream>>>(W1l, W1lT, FIN, D1);
  k_wT16<<<cdiv((long)FIN * D1, T), T, 0, stream>>>(W1r, W1rT, FIN, D1);
  k_wT16<<<cdiv((long)FIN * D1, T), T, 0, stream>>>(Ws1, Ws1T, FIN, D1);
  k_wT16<<<cdiv((long)D1 * OC, T), T, 0, stream>>>(W2l, W2lT, D1, OC);
  k_wT16<<<cdiv((long)D1 * OC, T), T, 0, stream>>>(W2r, W2rT, D1, OC);
  k_wT16<<<cdiv((long)D1 * OC, T), T, 0, stream>>>(Ws2, Ws2T, D1, OC);
  k_vadd<<<cdiv(D1, T), T, 0, stream>>>(bias1, bs1, cb1, D1);
  k_vadd<<<cdiv(OC, T), T, 0, stream>>>(bias2, bs2, cb2, OC);

  // -------- layer 1 --------
  int t1 = (NND / 32) * (D1 / 32);   // 32x32 macro-tiles per wave
  k_wmma_gemm<<<cdiv(t1, 8), T, 0, stream>>>(x16, W1lT, b1l, xl1, NND, D1, FIN, t1);
  k_wmma_gemm<<<cdiv(t1, 8), T, 0, stream>>>(x16, W1rT, b1r, xr1, NND, D1, FIN, t1);
  // skip + (bias1 + bskip1) written directly into the aggregation buffer
  k_wmma_gemm<<<cdiv(t1, 8), T, 0, stream>>>(x16, Ws1T, cb1, gat1, NND, D1, FIN, t1);

  k_fill_u32<<<cdiv((long)NND * H1N, T), T, 0, stream>>>(amax, NEG_INF_ORD, NND * H1N);
  k_fill_f32<<<cdiv((long)NND * H1N, T), T, 0, stream>>>(den, 0.0f, NND * H1N);
  k_edge_alpha<<<cdiv(NED, 8), T, 0, stream>>>(xl1, xr1, ea, We1, att1, src, dst,
                                               alph, amax, NED, H1N, C1);
  k_edge_exp<<<cdiv((long)NED * H1N, T), T, 0, stream>>>(alph, amax, dst, ex, den, NED, H1N);
  k_edge_scatter<<<cdiv(NED, 8), T, 0, stream>>>(xl1, ex, den, src, dst, gat1, NED, H1N, C1);
  k_ln_elu_h16<<<cdiv(NND, 8), T, 0, stream>>>(gat1, g1, be1, h16, NND, D1);

  // -------- layer 2 (heads=1) --------
  int t2 = (NND / 32) * (OC / 32);
  k_wmma_gemm<<<cdiv(t2, 8), T, 0, stream>>>(h16, W2lT, b2l, xl2, NND, OC, D1, t2);
  k_wmma_gemm<<<cdiv(t2, 8), T, 0, stream>>>(h16, W2rT, b2r, xr2, NND, OC, D1, t2);
  k_wmma_gemm<<<cdiv(t2, 8), T, 0, stream>>>(h16, Ws2T, cb2, gat2, NND, OC, D1, t2);

  k_fill_u32<<<cdiv(NND, T), T, 0, stream>>>(amax, NEG_INF_ORD, NND);
  k_fill_f32<<<cdiv(NND, T), T, 0, stream>>>(den, 0.0f, NND);
  k_edge_alpha<<<cdiv(NED, 8), T, 0, stream>>>(xl2, xr2, ea, We2, att2, src, dst,
                                               alph, amax, NED, 1, OC);
  k_edge_exp<<<cdiv(NED, T), T, 0, stream>>>(alph, amax, dst, ex, den, NED, 1);
  k_edge_scatter<<<cdiv(NED, 8), T, 0, stream>>>(xl2, ex, den, src, dst, gat2, NED, 1, OC);

  k_ln_out<<<cdiv(NND, 8), T, 0, stream>>>(gat2, g2, be2, (float*)d_out, NND, OC);
}
